// GETD_TN_43284680409623
// MI455X (gfx1250) — matrix operand order repaired and checked by reading, not compiled, not run on hardware
//
#include <hip/hip_runtime.h>

#define D_EE   100
#define RANKK  40
#define N_ENTT 10000
#define BB     128
#define EPSF   1e-5f

#define MDIM 10000   // (rho,x)
#define NDIM 10000   // (y,z)
#define KDIM 1600    // (a,c)
#define MPAD 10112   // 79 * 128
#define LDT  40      // LDS row stride in halfs (pad 32 -> 40 to spread banks)

typedef __attribute__((ext_vector_type(16))) _Float16 v16h;
typedef __attribute__((ext_vector_type(8)))  float    v8f;

union Frag { v16h h; uint4 u[2]; };

// ---------------------------------------------------------------------------
// Stage 1a: A2h[(rho*100+x), a*40+c] = sum_b f0[a,rho,b] * f1[b,x,c]   (f16)
// ---------------------------------------------------------------------------
__global__ __launch_bounds__(256) void make_A2(const float* __restrict__ f0,
                                               const float* __restrict__ f1,
                                               _Float16* __restrict__ A2h) {
  int rx = blockIdx.x;            // 0..9999
  int rho = rx / 100, x = rx % 100;
  __shared__ float s0[1600], s1[1600];
  for (int t = threadIdx.x; t < 1600; t += 256) {
    int p = t / 40, q = t % 40;
    s0[t] = f0[p * 4000 + rho * 40 + q];   // f0[a=p, rho, b=q]
    s1[t] = f1[p * 4000 + x   * 40 + q];   // f1[b=p, x,   c=q]
  }
  __syncthreads();
  for (int t = threadIdx.x; t < 1600; t += 256) {
    int a = t / 40, c = t % 40;
    float acc = 0.f;
    #pragma unroll 8
    for (int b = 0; b < 40; ++b) acc += s0[a * 40 + b] * s1[b * 40 + c];
    A2h[(size_t)rx * KDIM + t] = (_Float16)acc;
  }
}

// ---------------------------------------------------------------------------
// Stage 1b: BTh[(y*100+z), a*40+c] = sum_d f2[c,y,d] * f3[d,z,a]   (f16, B^T)
// ---------------------------------------------------------------------------
__global__ __launch_bounds__(256) void make_BT(const float* __restrict__ f2,
                                               const float* __restrict__ f3,
                                               _Float16* __restrict__ BTh) {
  int yz = blockIdx.x;            // 0..9999
  int y = yz / 100, z = yz % 100;
  __shared__ float s2[1600], s3[1600];
  for (int t = threadIdx.x; t < 1600; t += 256) {
    int p = t / 40, q = t % 40;
    s2[t] = f2[p * 4000 + y * 40 + q];     // f2[c=p, y, d=q]
    s3[t] = f3[p * 4000 + z * 40 + q];     // f3[d=p, z, a=q]
  }
  __syncthreads();
  for (int t = threadIdx.x; t < 1600; t += 256) {
    int a = t / 40, c = t % 40;
    float acc = 0.f;
    #pragma unroll 8
    for (int d = 0; d < 40; ++d) acc += s2[c * 40 + d] * s3[d * 40 + a];
    BTh[(size_t)yz * KDIM + t] = (_Float16)acc;
  }
}

// ---------------------------------------------------------------------------
// Stage 2: W = A2 @ B  via v_wmma_f32_16x16x32_f16.
// Block tile 128x128 (8 waves, wave tile 32x64, 2x4 WMMA frags).
// ---------------------------------------------------------------------------
__global__ __launch_bounds__(256) void tr_w_gemm(const _Float16* __restrict__ Ah,  // [MPAD,K]
                                                 const _Float16* __restrict__ Bt,  // [MPAD,K] = B^T
                                                 float* __restrict__ W) {          // [MDIM,NDIM]
  __shared__ _Float16 As[128 * LDT];
  __shared__ _Float16 Bs[128 * LDT];

  const int tid  = threadIdx.x;
  const int m0   = blockIdx.y * 128;
  const int n0   = blockIdx.x * 128;
  const int wave = tid >> 5;
  const int lane = tid & 31;
  const int rr   = lane & 15;
  const int hi   = lane >> 4;
  const int wm   = wave >> 1;   // 0..3 -> 32-row strips
  const int wn   = wave & 1;    // 0..1 -> 64-col strips

  v8f acc[2][4];
  #pragma unroll
  for (int i = 0; i < 2; ++i)
    #pragma unroll
    for (int j = 0; j < 4; ++j) acc[i][j] = {};

  for (int k0 = 0; k0 < KDIM; k0 += 32) {
    __syncthreads();
    #pragma unroll
    for (int it = 0; it < 2; ++it) {
      int id  = tid + it * 256;          // 0..511 -> 128 rows x 4 uint4
      int row = id >> 2;
      int cv  = (id & 3) * 8;
      *(uint4*)&As[row * LDT + cv] =
          *(const uint4*)(Ah + (size_t)(m0 + row) * KDIM + k0 + cv);
      *(uint4*)&Bs[row * LDT + cv] =
          *(const uint4*)(Bt + (size_t)(n0 + row) * KDIM + k0 + cv);
    }
    __syncthreads();

    Frag af[2], bf[4];
    #pragma unroll
    for (int fm = 0; fm < 2; ++fm) {
      int ar = (wm * 32 + fm * 16 + rr) * LDT;
      af[fm].u[0] = *(const uint4*)&As[ar + hi * 8];        // K = hi*8 .. +7
      af[fm].u[1] = *(const uint4*)&As[ar + 16 + hi * 8];   // K = 16+hi*8 .. +7
    }
    #pragma unroll
    for (int fn = 0; fn < 4; ++fn) {
      int br = (wn * 64 + fn * 16 + rr) * LDT;              // K = hi*8 .. +7, 16+hi*8 .. +7
      bf[fn].u[0] = *(const uint4*)&Bs[br + hi * 8];
      bf[fn].u[1] = *(const uint4*)&Bs[br + 16 + hi * 8];
    }
    #pragma unroll
    for (int fm = 0; fm < 2; ++fm)
      #pragma unroll
      for (int fn = 0; fn < 4; ++fn)
        acc[fm][fn] = __builtin_amdgcn_wmma_f32_16x16x32_f16(
            false, af[fm].h, false, bf[fn].h, (short)0, acc[fm][fn], false, false);
  }

  #pragma unroll
  for (int fm = 0; fm < 2; ++fm) {
    #pragma unroll
    for (int fn = 0; fn < 4; ++fn) {
      int col = n0 + wn * 64 + fn * 16 + rr;
      #pragma unroll
      for (int v = 0; v < 8; ++v) {
        int row = m0 + wm * 32 + fm * 16 + v + hi * 8;
        if (row < MDIM && col < NDIM)
          W[(size_t)row * NDIM + col] = acc[fm][fn][v];
      }
    }
  }
}

// ---------------------------------------------------------------------------
// BatchNorm over batch dim (biased variance), optional gather index.
// grid = D columns, block = 128 (= batch)
// ---------------------------------------------------------------------------
__global__ __launch_bounds__(128) void bn_kernel(const float* __restrict__ src,
                                                 const int* __restrict__ idx,
                                                 const float* __restrict__ gamma,
                                                 const float* __restrict__ beta,
                                                 float* __restrict__ out, int D) {
  int col = blockIdx.x;
  int b   = threadIdx.x;
  float x = idx ? src[(size_t)idx[b] * D + col] : src[(size_t)b * D + col];
  __shared__ float ssum[128], ssq[128];
  ssum[b] = x; ssq[b] = x * x;
  __syncthreads();
  for (int s = 64; s > 0; s >>= 1) {
    if (b < s) { ssum[b] += ssum[b + s]; ssq[b] += ssq[b + s]; }
    __syncthreads();
  }
  float mean = ssum[0] * (1.f / BB);
  float var  = ssq[0] * (1.f / BB) - mean * mean;
  out[(size_t)b * D + col] = gamma[col] * (x - mean) * rsqrtf(var + EPSF) + beta[col];
}

// ---------------------------------------------------------------------------
// P[b,t] = sum_{rho,x} r[b,rho]*e1[b,x] * A2h[(rho*100+x), t]
// ---------------------------------------------------------------------------
__global__ __launch_bounds__(256) void p_kernel(const _Float16* __restrict__ A2h,
                                                const float* __restrict__ rbn,
                                                const float* __restrict__ e1bn,
                                                float* __restrict__ P) {
  int b = blockIdx.x;
  __shared__ float rl[100], el[100];
  if (threadIdx.x < 100) {
    rl[threadIdx.x] = rbn[b * 100 + threadIdx.x];
    el[threadIdx.x] = e1bn[b * 100 + threadIdx.x];
  }
  __syncthreads();
  int t0 = threadIdx.x;
  float acc[7] = {0.f, 0.f, 0.f, 0.f, 0.f, 0.f, 0.f};
  for (int rho = 0; rho < 100; ++rho) {
    float rv = rl[rho];
    for (int x = 0; x < 100; ++x) {
      float u = rv * el[x];
      const _Float16* ap = A2h + (size_t)(rho * 100 + x) * KDIM;
      #pragma unroll
      for (int i = 0; i < 7; ++i) {
        int t = t0 + i * 256;
        if (t < KDIM) acc[i] += u * (float)ap[t];
      }
    }
  }
  #pragma unroll
  for (int i = 0; i < 7; ++i) {
    int t = t0 + i * 256;
    if (t < KDIM) P[(size_t)b * KDIM + t] = acc[i];
  }
}

// ---------------------------------------------------------------------------
// S[b,(y,z)] = sum_t BTh[(y,z),t] * P[b,t]    grid = (40 row-chunks, 128 b)
// ---------------------------------------------------------------------------
__global__ __launch_bounds__(256) void s_kernel(const _Float16* __restrict__ BTh,
                                                const float* __restrict__ P,
                                                float* __restrict__ S) {
  int b   = blockIdx.y;
  int row = blockIdx.x * 256 + threadIdx.x;
  __shared__ float pl[KDIM];
  for (int t = threadIdx.x; t < KDIM; t += 256) pl[t] = P[(size_t)b * KDIM + t];
  __syncthreads();
  if (row < NDIM) {
    const _Float16* bp = BTh + (size_t)row * KDIM;
    float acc = 0.f;
    for (int t = 0; t < KDIM; ++t) acc += pl[t] * (float)bp[t];
    S[(size_t)b * NDIM + row] = acc;
  }
}

// ---------------------------------------------------------------------------
// W_out[b,y] = sum_z S[b, y*100+z] * e2[b,z]
// ---------------------------------------------------------------------------
__global__ __launch_bounds__(128) void wout_kernel(const float* __restrict__ S,
                                                   const float* __restrict__ e2bn,
                                                   float* __restrict__ Wout) {
  int b = blockIdx.x;
  __shared__ float el[100];
  if (threadIdx.x < 100) el[threadIdx.x] = e2bn[b * 100 + threadIdx.x];
  __syncthreads();
  int y = threadIdx.x;
  if (y < 100) {
    const float* sp = S + (size_t)b * NDIM + y * 100;
    float acc = 0.f;
    #pragma unroll 4
    for (int z = 0; z < 100; ++z) acc += sp[z] * el[z];
    Wout[b * 100 + y] = acc;
  }
}

// ---------------------------------------------------------------------------
// scores = Wout_bn @ E^T ; softmax rows -> pred
// ---------------------------------------------------------------------------
__global__ __launch_bounds__(256) void score_softmax(const float* __restrict__ Wbn,
                                                     const float* __restrict__ E,
                                                     float* __restrict__ pred) {
  int b = blockIdx.x;
  __shared__ float wl[100];
  __shared__ float red[256];
  if (threadIdx.x < 100) wl[threadIdx.x] = Wbn[b * 100 + threadIdx.x];
  __syncthreads();
  float* out = pred + (size_t)b * N_ENTT;
  float lmax = -1e30f;
  for (int n = threadIdx.x; n < N_ENTT; n += 256) {
    const float* er = E + (size_t)n * 100;
    float s = 0.f;
    #pragma unroll 4
    for (int j = 0; j < 100; ++j) s += wl[j] * er[j];
    out[n] = s;
    lmax = fmaxf(lmax, s);
  }
  red[threadIdx.x] = lmax; __syncthreads();
  for (int s = 128; s > 0; s >>= 1) {
    if (threadIdx.x < s) red[threadIdx.x] = fmaxf(red[threadIdx.x], red[threadIdx.x + s]);
    __syncthreads();
  }
  float m = red[0];
  __syncthreads();
  float lsum = 0.f;
  for (int n = threadIdx.x; n < N_ENTT; n += 256) {
    float e = __expf(out[n] - m);
    out[n] = e;
    lsum += e;
  }
  red[threadIdx.x] = lsum; __syncthreads();
  for (int s = 128; s > 0; s >>= 1) {
    if (threadIdx.x < s) red[threadIdx.x] += red[threadIdx.x + s];
    __syncthreads();
  }
  float inv = 1.f / red[0];
  for (int n = threadIdx.x; n < N_ENTT; n += 256) out[n] *= inv;
}

// ---------------------------------------------------------------------------
extern "C" void kernel_launch(void* const* d_in, const int* in_sizes, int n_in,
                              void* d_out, int out_size, void* d_ws, size_t ws_size,
                              hipStream_t stream) {
  const float* E    = (const float*)d_in[0];
  const float* R    = (const float*)d_in[1];
  const float* f0   = (const float*)d_in[2];
  const float* f1   = (const float*)d_in[3];
  const float* f2   = (const float*)d_in[4];
  const float* f3   = (const float*)d_in[5];
  const float* bnrg = (const float*)d_in[6];
  const float* bnrb = (const float*)d_in[7];
  const float* bneg = (const float*)d_in[8];
  const float* bneb = (const float*)d_in[9];
  const float* bnwg = (const float*)d_in[10];
  const float* bnwb = (const float*)d_in[11];
  const int* r_idx  = (const int*)d_in[12];
  const int* e_idx1 = (const int*)d_in[13];
  const int* e_idx2 = (const int*)d_in[14];
  // d_in[15] = miss_ent_domain == 2 (middle entity dim), baked in.

  float* pred = (float*)d_out;                 // [128, 10000]
  float* W    = (float*)d_out + (size_t)BB * N_ENTT;  // [100,100,100,100]

  size_t off = 0;
  auto carve = [&](size_t bytes) -> void* {
    void* p = (char*)d_ws + off;
    off += (bytes + 255) & ~(size_t)255;
    return p;
  };
  _Float16* A2h  = (_Float16*)carve((size_t)MPAD * KDIM * sizeof(_Float16)); // 32.4 MB
  _Float16* BTh  = (_Float16*)carve((size_t)MPAD * KDIM * sizeof(_Float16)); // 32.4 MB
  float* rbn     = (float*)carve((size_t)BB * 100 * sizeof(float));
  float* e1bn    = (float*)carve((size_t)BB * 100 * sizeof(float));
  float* e2bn    = (float*)carve((size_t)BB * 100 * sizeof(float));
  float* P       = (float*)carve((size_t)BB * KDIM * sizeof(float));
  float* S       = (float*)carve((size_t)BB * NDIM * sizeof(float));          // 5.1 MB
  float* Wout    = (float*)carve((size_t)BB * 100 * sizeof(float));
  float* Woutbn  = (float*)carve((size_t)BB * 100 * sizeof(float));
  (void)ws_size; (void)in_sizes; (void)n_in; (void)out_size;

  // BN of gathered embeddings
  bn_kernel<<<100, 128, 0, stream>>>(R, r_idx,  bnrg, bnrb, rbn,  100);
  bn_kernel<<<100, 128, 0, stream>>>(E, e_idx1, bneg, bneb, e1bn, 100);
  bn_kernel<<<100, 128, 0, stream>>>(E, e_idx2, bneg, bneb, e2bn, 100);

  // TR factors -> GEMM operands (f16)
  make_A2<<<10000, 256, 0, stream>>>(f0, f1, A2h);
  make_BT<<<10000, 256, 0, stream>>>(f2, f3, BTh);

  // W = A2 @ B  (10000 x 1600 x 10000, WMMA f16->f32)
  tr_w_gemm<<<dim3(79, 79), 256, 0, stream>>>(A2h, BTh, W);

  // Scoring path via ring factorization (avoids r @ W)
  p_kernel<<<BB, 256, 0, stream>>>(A2h, rbn, e1bn, P);
  s_kernel<<<dim3(40, BB), 256, 0, stream>>>(BTh, P, S);
  wout_kernel<<<BB, 128, 0, stream>>>(S, e2bn, Wout);
  bn_kernel<<<100, 128, 0, stream>>>(Wout, nullptr, bnwg, bnwb, Woutbn, 100);
  score_softmax<<<BB, 256, 0, stream>>>(Woutbn, E, pred);
}
